// AVWGCN_49460843381226
// MI455X (gfx1250) — matrix operand order repaired and checked by reading, not compiled, not run on hardware
//
#include <hip/hip_runtime.h>
#include <hip/hip_bf16.h>

// ---------- types ----------
typedef __attribute__((ext_vector_type(16))) __bf16 v16bf;
typedef __attribute__((ext_vector_type(8)))  float  v8f;
typedef __attribute__((ext_vector_type(4)))  unsigned int u32x4;
typedef __attribute__((ext_vector_type(4)))  unsigned int v4u;
typedef __attribute__((ext_vector_type(8)))  int v8i;
typedef __attribute__((ext_vector_type(4)))  int v4i;

union AFrag { u32x4 q[2]; v16bf v; };

__device__ __forceinline__ unsigned short f2bf(float f) {
    union { float f; unsigned int u; } x; x.f = f;
    unsigned int r = x.u + 0x7FFFu + ((x.u >> 16) & 1u);   // round-to-nearest-even
    return (unsigned short)(r >> 16);
}

// Problem constants
#define NN   2048          // nodes
#define EMB  16
#define CIN  32
#define COUT 64
#define BT   512           // B*T
#define NJ   (BT*CIN)      // 16384 wide GEMM N-dim
#define KW   96            // 3*CIN

#if __has_builtin(__builtin_amdgcn_tensor_load_to_lds) && __has_builtin(__builtin_amdgcn_s_wait_tensorcnt)
#define HAVE_TDM 1
#else
#define HAVE_TDM 0
#endif

#if HAVE_TDM
// Generic pointers into LDS carry the 32-bit LDS offset in their low bits.
__device__ __forceinline__ unsigned lds_addr_of(const void* p) {
    return (unsigned)(unsigned long long)p;
}

// Issue one TDM 2-D tile load (bf16 elements) per the CDNA5 D# layout.
//   tile_d1 rows of tile_d0 elements, global row stride row_stride (elements),
//   optional LDS padding: pad codes per ISA (interval 2^(c+1) DW, amount c+1 DW).
__device__ __forceinline__ void tdm_load_2d(unsigned lds_byte_addr, const void* gptr,
                                            unsigned tile_d0, unsigned tile_d1,
                                            unsigned long long row_stride,
                                            int pad_en, int pad_interval_c, int pad_amount_c) {
    unsigned long long ga = (unsigned long long)gptr;
    v4u g0;
    g0[0] = 1u;                                    // count=1, user mode, no gather
    g0[1] = lds_byte_addr;                         // D#.lds_addr
    g0[2] = (unsigned)(ga & 0xFFFFFFFFu);          // D#.global_addr[31:0]
    g0[3] = (unsigned)((ga >> 32) & 0x01FFFFFFu) | (2u << 30);  // addr[56:32] | type=2
    unsigned d0 = 0x00010000u;                     // data_size=1 (2 bytes), mask=0
    if (pad_en)
        d0 |= (1u << 20) | ((unsigned)pad_interval_c << 22) | ((unsigned)pad_amount_c << 25);
    v8i g1;
    g1[0] = (int)d0;
    g1[1] = (int)((tile_d0 & 0xFFFFu) << 16);      // tensor_dim0 = tile_d0 (lo16 @ bits 63:48)
    g1[2] = (int)(((tile_d0 >> 16) & 0xFFFFu) | ((tile_d1 & 0xFFFFu) << 16)); // dim0 hi | dim1 lo
    g1[3] = (int)(((tile_d1 >> 16) & 0xFFFFu) | ((tile_d0 & 0xFFFFu) << 16)); // dim1 hi | tile_dim0
    g1[4] = (int)(tile_d1 & 0xFFFFu);              // tile_dim1 | tile_dim2=0
    g1[5] = (int)(unsigned)(row_stride & 0xFFFFFFFFu);          // tensor_dim0_stride lo32
    g1[6] = (int)(unsigned)((row_stride >> 32) & 0xFFFFu);      // stride hi16, dim1_stride=0
    g1[7] = 0;
    v4i z4 = {0, 0, 0, 0};
    v8i z8 = {0, 0, 0, 0, 0, 0, 0, 0};
    __builtin_amdgcn_tensor_load_to_lds(g0, g1, z4, z4, z8, 0);
}
#endif

// ---------------------------------------------------------------------------
// Kernel 1: x[bt,n,c] (f32) -> Xr[n, bt*32+c] (bf16).  Both sides coalesced.
// ---------------------------------------------------------------------------
__global__ void xpose_kernel(const float* __restrict__ x, unsigned short* __restrict__ Xr) {
    size_t idx = (size_t)blockIdx.x * 256 + threadIdx.x;   // flat x index
    int c  = (int)(idx & 31);
    size_t r = idx >> 5;            // bt*2048 + n
    int n  = (int)(r & 2047);
    int bt = (int)(r >> 11);
    Xr[(size_t)n * NJ + (size_t)bt * 32 + c] = f2bf(x[idx]);
}

// ---------------------------------------------------------------------------
// Kernel 2: S = softmax(relu(E E^T)) rows, written bf16 row-major.
// ---------------------------------------------------------------------------
__global__ void supports_kernel(const float* __restrict__ E, unsigned short* __restrict__ S) {
    int n = blockIdx.x, tid = threadIdx.x;
    float En[EMB];
#pragma unroll
    for (int d = 0; d < EMB; d++) En[d] = E[n * EMB + d];

    float vals[8];
    float vmax = 0.f;
#pragma unroll
    for (int it = 0; it < 8; it++) {
        int m = tid + it * 256;
        const float* Em = E + m * EMB;
        float dot = 0.f;
#pragma unroll
        for (int d = 0; d < EMB; d++) dot += Em[d] * En[d];
        dot = fmaxf(dot, 0.f);               // relu
        vals[it] = dot;
        vmax = fmaxf(vmax, dot);
    }
    __shared__ float red[256];
    red[tid] = vmax; __syncthreads();
    for (int s = 128; s > 0; s >>= 1) { if (tid < s) red[tid] = fmaxf(red[tid], red[tid + s]); __syncthreads(); }
    float bmax = red[0]; __syncthreads();

    float lsum = 0.f;
#pragma unroll
    for (int it = 0; it < 8; it++) { vals[it] = __expf(vals[it] - bmax); lsum += vals[it]; }
    red[tid] = lsum; __syncthreads();
    for (int s = 128; s > 0; s >>= 1) { if (tid < s) red[tid] += red[tid + s]; __syncthreads(); }
    float inv = 1.0f / red[0];
#pragma unroll
    for (int it = 0; it < 8; it++)
        S[(size_t)n * NN + tid + it * 256] = f2bf(vals[it] * inv);
}

// ---------------------------------------------------------------------------
// Kernel 3: Wt[n][o][k*32+i] = sum_d E[n,d]*Wp[d,k,i,o];  bias[n][o] = E[n]·Bp[:,o]
// ---------------------------------------------------------------------------
__global__ void wprep_kernel(const float* __restrict__ E, const float* __restrict__ Wp,
                             const float* __restrict__ Bp, unsigned short* __restrict__ Wt,
                             float* __restrict__ biasW) {
    int n = blockIdx.x, tid = threadIdx.x;
    float En[EMB];
#pragma unroll
    for (int d = 0; d < EMB; d++) En[d] = E[n * EMB + d];

#pragma unroll 4
    for (int j = 0; j < 24; j++) {
        int f = tid + j * 256;              // 0..6143 = o*96 + kidx
        int o = f / KW, kidx = f % KW;
        int k = kidx >> 5, i = kidx & 31;
        float acc = 0.f;
#pragma unroll
        for (int d = 0; d < EMB; d++)
            acc += En[d] * Wp[(((size_t)d * 3 + k) * 32 + i) * 64 + o];
        Wt[(size_t)n * (COUT * KW) + f] = f2bf(acc);
    }
    if (tid < COUT) {
        float b = 0.f;
#pragma unroll
        for (int d = 0; d < EMB; d++) b += En[d] * Bp[d * COUT + tid];
        biasW[n * COUT + tid] = b;
    }
}

// ---------------------------------------------------------------------------
// Kernel 4: bf16 GEMM, C = scale*(A@B) + diagAdd*I (bf16 out, f32 accum)
// A [M,K] rm, B [K,Nc] rm (transposed into LDS), 128x128x32 double-buffered.
// A tile staged by the Tensor Data Mover (hardware row padding to 40 elems).
// 256 threads = 8 waves (4M x 2N); wave tile 32x64 -> 8 wmma per k-step.
// ---------------------------------------------------------------------------
__global__ __launch_bounds__(256) void gemm_bf16_kernel(
    const unsigned short* __restrict__ A, const unsigned short* __restrict__ B,
    unsigned short* __restrict__ C, int M, int Nc, int K, float scale, float diagAdd) {

    __shared__ unsigned short As[2][128 * 40];   // [row][k], 80B rows (TDM-padded)
    __shared__ unsigned short Bs[2][128 * 40];   // [col][k] (transposed during stage)

    const int tid  = threadIdx.x;
    const int m0   = blockIdx.y * 128;
    const int n0   = blockIdx.x * 128;
    const int nk   = K >> 5;

    // B staging: thread handles k-row pair {bkp,bkp+1}, 8 columns; packs pairs -> b32.
    const int bkp = (tid >> 4) << 1;          // 0,2,..,30
    const int bcc = (tid & 15) * 8;           // column chunk of 8
#if !HAVE_TDM
    const int arow = tid >> 1, ahalf = (tid & 1) * 16;
#endif

    auto stageA = [&](int buf, int kt) {
#if HAVE_TDM
        if (tid == 0)
            tdm_load_2d(lds_addr_of(&As[buf][0]), A + (size_t)m0 * K + (kt << 5),
                        /*tile_d0=*/32, /*tile_d1=*/128, /*row_stride=*/(unsigned long long)K,
                        /*pad_en=*/1, /*interval 16 DW=*/3, /*amount 4 DW=*/3);
#else
        const u32x4* as = (const u32x4*)(A + (size_t)(m0 + arow) * K + (kt << 5) + ahalf);
        u32x4* ad = (u32x4*)&As[buf][arow * 40 + ahalf];
        ad[0] = as[0]; ad[1] = as[1];
#endif
    };
    auto stageB = [&](int buf, int kt) {
        int k0 = kt << 5;
        union { u32x4 q; unsigned short s[8]; } r0, r1;
        r0.q = *(const u32x4*)(B + (size_t)(k0 + bkp) * Nc + n0 + bcc);
        r1.q = *(const u32x4*)(B + (size_t)(k0 + bkp + 1) * Nc + n0 + bcc);
        unsigned* dst = (unsigned*)&Bs[buf][0];
#pragma unroll
        for (int e = 0; e < 8; e++)
            dst[(bcc + e) * 20 + (bkp >> 1)] = (unsigned)r0.s[e] | ((unsigned)r1.s[e] << 16);
    };

    v8f acc[2][4];
#pragma unroll
    for (int fm = 0; fm < 2; fm++)
#pragma unroll
        for (int fn = 0; fn < 4; fn++)
#pragma unroll
            for (int r = 0; r < 8; r++) acc[fm][fn][r] = 0.f;

    const int lane = tid & 31, wid = tid >> 5;
    const int wm = wid & 3, wn = wid >> 2;
    const int l15 = lane & 15;
    const int a0 = (lane & 16) ? 8 : 0;     // A frag: K chunks {a0, a0+16}
    const int b0 = (lane & 16) ? 8 : 0;     // B frag: K chunks {b0, b0+16}

    stageB(0, 0);
    stageA(0, 0);
#if HAVE_TDM
    __builtin_amdgcn_s_wait_tensorcnt(0);
#endif
    __syncthreads();

    for (int kt = 0; kt < nk; kt++) {
        int cur = kt & 1;
        if (kt + 1 < nk) { stageB(cur ^ 1, kt + 1); stageA(cur ^ 1, kt + 1); }
        if (kt + 2 < nk)
            __builtin_prefetch(B + (size_t)(((kt + 2) << 5) + bkp) * Nc + n0 + bcc, 0, 1);

        AFrag af[2], bfr[4];
#pragma unroll
        for (int fm = 0; fm < 2; fm++) {
            int row = wm * 32 + fm * 16 + l15;
            const unsigned short* p = &As[cur][row * 40 + a0];
            af[fm].q[0] = *(const u32x4*)(p);
            af[fm].q[1] = *(const u32x4*)(p + 16);
        }
#pragma unroll
        for (int fn = 0; fn < 4; fn++) {
            int col = wn * 64 + fn * 16 + l15;
            const unsigned short* p = &Bs[cur][col * 40 + b0];
            bfr[fn].q[0] = *(const u32x4*)(p);
            bfr[fn].q[1] = *(const u32x4*)(p + 16);
        }
#pragma unroll
        for (int fm = 0; fm < 2; fm++)
#pragma unroll
            for (int fn = 0; fn < 4; fn++)
                acc[fm][fn] = __builtin_amdgcn_wmma_f32_16x16x32_bf16(
                    false, af[fm].v, false, bfr[fn].v, (short)0, acc[fm][fn], false, false);
#if HAVE_TDM
        __builtin_amdgcn_s_wait_tensorcnt(0);   // next A tile landed (wave 0)
#endif
        __syncthreads();
    }

    // Epilogue: C/D layout — VGPR r, lane l -> row r + 8*(l>=16), col l%16
#pragma unroll
    for (int fm = 0; fm < 2; fm++)
#pragma unroll
        for (int fn = 0; fn < 4; fn++) {
            int col   = n0 + wn * 64 + fn * 16 + l15;
            int rbase = m0 + wm * 32 + fm * 16 + ((lane >> 4) << 3);
#pragma unroll
            for (int r = 0; r < 8; r++) {
                int row = rbase + r;
                float v = acc[fm][fn][r] * scale + ((row == col) ? diagAdd : 0.f);
                C[(size_t)row * Nc + col] = f2bf(v);
            }
        }
}

// ---------------------------------------------------------------------------
// Kernel 5: out[bt,n,o] = [Xr|Y1|Y2]_n (512x96) @ Wt_n^T (96x64) + bias[n]
// One node per block, 128 bt-rows (8 waves x 16); Wt_n TDM-staged into LDS.
// ---------------------------------------------------------------------------
__global__ __launch_bounds__(256) void out_kernel(
    const unsigned short* __restrict__ Xr, const unsigned short* __restrict__ Y1,
    const unsigned short* __restrict__ Y2, const unsigned short* __restrict__ Wt,
    const float* __restrict__ biasW, float* __restrict__ out) {

    __shared__ unsigned short Ws[COUT * KW];   // Wt_n [o][kidx], stride 96

    const int n   = blockIdx.x;
    const int bt0 = blockIdx.y * 128;
    const int tid = threadIdx.x;

#if HAVE_TDM
    if (tid == 0)
        tdm_load_2d(lds_addr_of(&Ws[0]), Wt + (size_t)n * (COUT * KW),
                    /*tile_d0=*/COUT * KW, /*tile_d1=*/1, /*row_stride=*/COUT * KW, 0, 0, 0);
    __builtin_amdgcn_s_wait_tensorcnt(0);
#else
    {   // cooperative copy fallback: 24 elems / thread
        int row = tid >> 2, ch = (tid & 3) * 24;
        const u32x4* src = (const u32x4*)(Wt + (size_t)n * (COUT * KW) + row * KW + ch);
        u32x4* dst = (u32x4*)&Ws[row * KW + ch];
        dst[0] = src[0]; dst[1] = src[1]; dst[2] = src[2];
    }
#endif
    __syncthreads();

    const int lane = tid & 31, wid = tid >> 5;
    const int l15 = lane & 15;
    const int a0 = (lane & 16) ? 8 : 0;
    const int b0 = (lane & 16) ? 8 : 0;
    const int m0 = bt0 + wid * 16;

    v8f acc[4];
#pragma unroll
    for (int fn = 0; fn < 4; fn++) {
        float bv = biasW[n * COUT + fn * 16 + l15];   // bias depends on column only
#pragma unroll
        for (int r = 0; r < 8; r++) acc[fn][r] = bv;
    }

    const unsigned short* bases[3] = {
        Xr + (size_t)n * NJ, Y1 + (size_t)n * NJ, Y2 + (size_t)n * NJ };

#pragma unroll
    for (int s = 0; s < 3; s++) {
        const unsigned short* base = bases[s];
        AFrag af;
        int m = m0 + l15;
        const unsigned short* ap = base + (size_t)m * 32 + a0;
        af.q[0] = *(const u32x4*)(ap);
        af.q[1] = *(const u32x4*)(ap + 16);
#pragma unroll
        for (int fn = 0; fn < 4; fn++) {
            AFrag bf_;
            const unsigned short* bp = &Ws[(fn * 16 + l15) * KW + s * 32 + b0];
            bf_.q[0] = *(const u32x4*)(bp);
            bf_.q[1] = *(const u32x4*)(bp + 16);
            acc[fn] = __builtin_amdgcn_wmma_f32_16x16x32_bf16(
                false, af.v, false, bf_.v, (short)0, acc[fn], false, false);
        }
    }

#pragma unroll
    for (int fn = 0; fn < 4; fn++) {
        int o = fn * 16 + l15;
        int btbase = m0 + ((lane >> 4) << 3);
#pragma unroll
        for (int r = 0; r < 8; r++)
            out[((size_t)(btbase + r) * NN + n) * COUT + o] = acc[fn][r];
    }
}

// ---------------------------------------------------------------------------
extern "C" void kernel_launch(void* const* d_in, const int* in_sizes, int n_in,
                              void* d_out, int out_size, void* d_ws, size_t ws_size,
                              hipStream_t stream) {
    const float* x  = (const float*)d_in[0];   // [16,32,2048,32]
    const float* E  = (const float*)d_in[1];   // [2048,16]
    const float* Wp = (const float*)d_in[2];   // [16,3,32,64]
    const float* Bp = (const float*)d_in[3];   // [16,64]
    float* out = (float*)d_out;                // [16,32,2048,64]

    char* ws = (char*)d_ws;
    size_t off = 0;
    auto carve = [&](size_t bytes) -> char* {
        char* p = ws + off; off += (bytes + 255) & ~(size_t)255; return p;
    };
    unsigned short* S   = (unsigned short*)carve((size_t)NN * NN * 2);   //  8.4 MB
    unsigned short* T2  = (unsigned short*)carve((size_t)NN * NN * 2);   //  8.4 MB
    unsigned short* Xr  = (unsigned short*)carve((size_t)NN * NJ * 2);   // 67 MB
    unsigned short* Y1  = (unsigned short*)carve((size_t)NN * NJ * 2);   // 67 MB
    unsigned short* Y2  = (unsigned short*)carve((size_t)NN * NJ * 2);   // 67 MB
    unsigned short* Wt  = (unsigned short*)carve((size_t)NN * COUT * KW * 2); // 25 MB
    float*          bw  = (float*)carve((size_t)NN * COUT * 4);          // 0.5 MB

    // 1) reshape/convert x -> Xr bf16
    xpose_kernel<<<(BT * NN * CIN) / 256, 256, 0, stream>>>(x, Xr);
    // 2) adaptive adjacency S
    supports_kernel<<<NN, 256, 0, stream>>>(E, S);
    // 3) node-adaptive weights / bias
    wprep_kernel<<<NN, 256, 0, stream>>>(E, Wp, Bp, Wt, bw);
    // 4) Chebyshev T2 = 2*S@S - I  (2048^3)
    gemm_bf16_kernel<<<dim3(NN / 128, NN / 128), 256, 0, stream>>>(S, S, T2, NN, NN, NN, 2.0f, -1.0f);
    // 5) hot GEMMs: Y1 = S@Xr, Y2 = T2@Xr  (2048 x 16384 x 2048 each)
    gemm_bf16_kernel<<<dim3(NJ / 128, NN / 128), 256, 0, stream>>>(S,  Xr, Y1, NN, NJ, NN, 1.0f, 0.0f);
    gemm_bf16_kernel<<<dim3(NJ / 128, NN / 128), 256, 0, stream>>>(T2, Xr, Y2, NN, NJ, NN, 1.0f, 0.0f);
    // 6) per-node output contraction + bias
    out_kernel<<<dim3(NN, BT / 128), 256, 0, stream>>>(Xr, Y1, Y2, Wt, bw, out);
}
